// FFT_MLP_KAN_v1_81913616269803
// MI455X (gfx1250) — matrix operand order repaired and checked by reading, not compiled, not run on hardware
//
#include <hip/hip_runtime.h>
#include <hip/hip_bf16.h>
#include <math.h>

typedef __attribute__((ext_vector_type(16))) __bf16 v16bf;
typedef __attribute__((ext_vector_type(8)))  float  v8f;

// ---------------------------------------------------------------------------
// FFT feature kernel: x (B,64,14) flat -> reshape(B,14,64); per (b,c) two
// 32-sample windows, keep 9 rfft bins: [abs(prev), ang(cur), abs(cur), ang(cur)]
// ---------------------------------------------------------------------------
__global__ void fft_feat_kernel(const float* __restrict__ x,
                                float* __restrict__ h0, int B) {
    int idx = blockIdx.x * blockDim.x + threadIdx.x;
    if (idx >= B * 14) return;
    int b = idx / 14, c = idx % 14;
    const float* src = x + (size_t)b * 896 + c * 64;
    float w0[32], w1[32];
#pragma unroll
    for (int n = 0; n < 32; ++n) { w0[n] = src[n]; w1[n] = src[32 + n]; }
    float* dst = h0 + (size_t)b * 504 + c * 36;
#pragma unroll 1
    for (int k = 0; k < 9; ++k) {
        float re0 = 0.f, im0 = 0.f, re1 = 0.f, im1 = 0.f;
        float w = -0.19634954084936207f * (float)k;   // -2*pi*k/32
#pragma unroll
        for (int n = 0; n < 32; ++n) {
            float sn, cn;
            __sincosf(w * (float)n, &sn, &cn);
            re0 += w0[n] * cn; im0 += w0[n] * sn;
            re1 += w1[n] * cn; im1 += w1[n] * sn;
        }
        float ab0 = sqrtf(re0 * re0 + im0 * im0);
        float ab1 = sqrtf(re1 * re1 + im1 * im1);
        float an1 = atan2f(im1, re1);
        dst[k]      = ab0;
        dst[9 + k]  = an1;   // reference uses cur-window phase twice
        dst[18 + k] = ab1;
        dst[27 + k] = an1;
    }
}

// ---------------------------------------------------------------------------
// Pack combined bf16 weights: Wg[o][i*14 + 0] = base_w[o][i]
//                             Wg[o][i*14+1+c] = spline_w[o][i][c]*scaler[o][i]
// Zero-padded to (out_p, inp*14); inp = ceil(in/32)*32, out_p = ceil(out/16)*16
// ---------------------------------------------------------------------------
__global__ void pack_w_kernel(const float* __restrict__ base_w,
                              const float* __restrict__ spline_w,
                              const float* __restrict__ scaler,
                              __bf16* __restrict__ Wg,
                              int out, int out_p, int in, int inp) {
    int idx = blockIdx.x * blockDim.x + threadIdx.x;
    if (idx >= out_p * inp) return;
    int o = idx / inp, i = idx % inp;
    __bf16* dst = Wg + ((size_t)o * inp + i) * 14;
    if (o < out && i < in) {
        float bw = base_w[o * in + i];
        float sc = scaler[o * in + i];
        dst[0] = (__bf16)bw;
#pragma unroll
        for (int c = 0; c < 13; ++c)
            dst[1 + c] = (__bf16)(spline_w[((size_t)o * in + i) * 13 + c] * sc);
    } else {
#pragma unroll
        for (int s = 0; s < 14; ++s) dst[s] = (__bf16)0.0f;
    }
}

// ---------------------------------------------------------------------------
// Fused KAN layer: Hout(B,out) = expand(Hin)(B,inp*14) @ Wg(out_p,inp*14)^T
//
// Workgroup: 32 rows x out_p cols; one wave per 16-col tile with TWO
// independent accumulator chains (rows 0-15 and 16-31) -> better XDL
// pipelining + 2x weight reuse per b-fragment.
//
// Expansion uses the closed-form uniform cubic B-spline (only 4 nonzero
// bases): zero the LDS tile with vector stores, then scatter silu + 4 spline
// values per feature.  ~5x fewer VALU ops than the Cox-de Boor recursion.
//
// Lane layouts per CDNA5 ISA 7.12.2 (wave32).
// ---------------------------------------------------------------------------
template <int OUTP>
__global__ __launch_bounds__(2 * OUTP)
void kan_gemm_kernel(const float* __restrict__ Hin, int in, int inp,
                     const __bf16* __restrict__ Wg,
                     float* __restrict__ Hout, int out) {
    constexpr int NW  = OUTP / 16;
    constexpr int TPB = 32 * NW;
    const int Kp = inp * 14;
    __shared__ __align__(16) __bf16 smA[32][456];   // 32 x 448 (+8 pad)

    const int tid  = threadIdx.x;
    const int wave = tid >> 5;
    const int lane = tid & 31;
    const int m    = lane & 15;               // row-in-tile (A) / col (B,D)
    const int g    = lane >> 4;               // half-wave selector
    const int row0 = blockIdx.x * 32;

    v8f acc0 = {};
    v8f acc1 = {};

    const int nch = inp >> 5;                 // chunks of 32 input features
    for (int ch = 0; ch < nch; ++ch) {
        const int f0 = ch << 5;

        // ---- zero the LDS tile (vectorized) ----
        uint4* z = (uint4*)&smA[0][0];
        constexpr int NZ = (32 * 456 * 2) / 16;   // 1824 uint4
        for (int idx = tid; idx < NZ; idx += TPB) z[idx] = uint4{0u, 0u, 0u, 0u};
        __syncthreads();

        // ---- scatter expansion: silu + 4 nonzero cubic B-spline bases ----
        for (int idx = tid; idx < 1024; idx += TPB) {
            int r = idx >> 5, fl = idx & 31;
            int i = f0 + fl;
            if (i < in) {
                float x = Hin[(row0 + r) * in + i];
                __bf16* dst = &smA[r][fl * 14];
                dst[0] = (__bf16)(x / (1.0f + __expf(-x)));       // silu
                float t = (x + 0.3f) * 10.0f;                     // knot coord
                if (t >= 0.0f && t < 16.0f) {
                    int   iv = (int)t;                            // interval 0..15
                    float u  = t - (float)iv;
                    float u2 = u * u, u3 = u2 * u, um = 1.0f - u;
                    float S0 = u3 * (1.0f / 6.0f);
                    float S1 = (-3.0f * u3 + 3.0f * u2 + 3.0f * u + 1.0f) * (1.0f / 6.0f);
                    float S2 = (3.0f * u3 - 6.0f * u2 + 4.0f) * (1.0f / 6.0f);
                    float S3 = um * um * um * (1.0f / 6.0f);
                    // basis c = iv - s gets S_s, slot = 1 + c, c clipped to [0,12]
                    if (iv <= 12)            dst[1 + iv] = (__bf16)S0;
                    if (iv >= 1 && iv <= 13) dst[iv]     = (__bf16)S1;
                    if (iv >= 2 && iv <= 14) dst[iv - 1] = (__bf16)S2;
                    if (iv >= 3)             dst[iv - 2] = (__bf16)S3;
                }
            }
        }
        __syncthreads();

        // ---- WMMA over this chunk's 448 K values ----
        const __bf16* wrow = Wg + (size_t)(wave * 16 + m) * Kp + (size_t)f0 * 14;
        __builtin_prefetch(wrow + 448, 0, 1);       // next chunk's weight row
#pragma unroll
        for (int kk = 0; kk < 448; kk += 32) {
            v16bf a0, a1, b;
            // A 16x32 bf16 layout: VGPR j holds K = (j/4)*16 + g*8 + (j%4)*2 (+1)
#pragma unroll
            for (int j = 0; j < 8; ++j) {
                int kb = kk + ((j >> 2) << 4) + (g << 3) + ((j & 3) << 1);
                a0[2 * j]     = smA[m][kb];
                a0[2 * j + 1] = smA[m][kb + 1];
                a1[2 * j]     = smA[16 + m][kb];
                a1[2 * j + 1] = smA[16 + m][kb + 1];
            }
            // B 32x16 bf16: element t holds K = (t/8)*16 + g*8 + (t%8) at column m
#pragma unroll
            for (int t = 0; t < 16; ++t)
                b[t] = wrow[kk + ((t >> 3) << 4) + (g << 3) + (t & 7)];
            acc0 = __builtin_amdgcn_wmma_f32_16x16x32_bf16(
                false, a0, false, b, (short)0, acc0, false, false);
            acc1 = __builtin_amdgcn_wmma_f32_16x16x32_bf16(
                false, a1, false, b, (short)0, acc1, false, false);
        }
        __syncthreads();
    }

    // D 16x16 f32 layout: VGPR v -> M = v + 8*g, N = m
    const int col = wave * 16 + m;
    if (col < out) {
#pragma unroll
        for (int v = 0; v < 8; ++v) {
            int M = v + 8 * g;
            Hout[(size_t)(row0 + M) * out + col]      = acc0[v];
            Hout[(size_t)(row0 + 16 + M) * out + col] = acc1[v];
        }
    }
}

// ---------------------------------------------------------------------------
// 3 tiny MLP heads per sample: 40->40 (identity act) ->20 (leaky 0.05) ->1 ->sigmoid
// ---------------------------------------------------------------------------
__global__ void heads_kernel(const float* __restrict__ h,
                             const float* __restrict__ W1, const float* __restrict__ b1,
                             const float* __restrict__ W2, const float* __restrict__ b2,
                             const float* __restrict__ W3, const float* __restrict__ b3,
                             float* __restrict__ out, int B) {
    int bidx = blockIdx.x * blockDim.x + threadIdx.x;
    if (bidx >= B) return;
    float hv[40];
#pragma unroll
    for (int i = 0; i < 40; ++i) hv[i] = h[(size_t)bidx * 40 + i];
#pragma unroll 1
    for (int head = 0; head < 3; ++head) {
        const float* w1 = W1 + head * 40 * 40;
        const float* B1 = b1 + head * 40;
        const float* w2 = W2 + head * 20 * 40;
        const float* B2 = b2 + head * 20;
        const float* w3 = W3 + head * 20;
        float y1[40];
        for (int o = 0; o < 40; ++o) {
            float s = B1[o];
            for (int i = 0; i < 40; ++i) s += w1[o * 40 + i] * hv[i];
            y1[o] = s;                                // LeakyReLU(1.0) == identity
        }
        float y2[20];
        for (int o = 0; o < 20; ++o) {
            float s = B2[o];
            for (int i = 0; i < 40; ++i) s += w2[o * 40 + i] * y1[i];
            y2[o] = (s >= 0.f) ? s : 0.05f * s;       // LeakyReLU(0.05)
        }
        float s = b3[head];
        for (int i = 0; i < 20; ++i) s += w3[i] * y2[i];
        out[(size_t)bidx * 3 + head] = 1.0f / (1.0f + __expf(-s));
    }
}

// ---------------------------------------------------------------------------
extern "C" void kernel_launch(void* const* d_in, const int* in_sizes, int n_in,
                              void* d_out, int out_size, void* d_ws, size_t ws_size,
                              hipStream_t stream) {
    (void)n_in; (void)out_size; (void)ws_size;
    const float* x        = (const float*)d_in[0];
    const float* k1_base  = (const float*)d_in[1];
    const float* k1_spl   = (const float*)d_in[2];
    const float* k1_scl   = (const float*)d_in[3];
    const float* k2_base  = (const float*)d_in[4];
    const float* k2_spl   = (const float*)d_in[5];
    const float* k2_scl   = (const float*)d_in[6];
    const float* k3_base  = (const float*)d_in[7];
    const float* k3_spl   = (const float*)d_in[8];
    const float* k3_scl   = (const float*)d_in[9];
    const float* k4_base  = (const float*)d_in[10];
    const float* k4_spl   = (const float*)d_in[11];
    const float* k4_scl   = (const float*)d_in[12];
    const float* hW1      = (const float*)d_in[13];
    const float* hb1      = (const float*)d_in[14];
    const float* hW2      = (const float*)d_in[15];
    const float* hb2      = (const float*)d_in[16];
    const float* hW3      = (const float*)d_in[17];
    const float* hb3      = (const float*)d_in[18];
    float* out = (float*)d_out;

    const int B = in_sizes[0] / (64 * 14);    // 8192

    // workspace carve-out
    char* ws = (char*)d_ws;
    size_t off = 0;
    auto carve = [&](size_t bytes) -> void* {
        off = (off + 255) & ~(size_t)255;
        void* p = ws + off;
        off += bytes;
        return p;
    };
    float*  h0  = (float*)carve((size_t)B * 504 * sizeof(float));
    float*  hA  = (float*)carve((size_t)B * 160 * sizeof(float));
    float*  hB  = (float*)carve((size_t)B * 160 * sizeof(float));
    // layer dims: (in, inp, out, out_p)
    // L1: 504 -> 512,  80 ->  80 ; L2: 80 -> 96, 160 -> 160
    // L3: 160 -> 160,  80 ->  80 ; L4: 80 -> 96,  40 ->  48
    __bf16* Wg1 = (__bf16*)carve((size_t)80  * 512 * 14 * sizeof(__bf16));
    __bf16* Wg2 = (__bf16*)carve((size_t)160 * 96  * 14 * sizeof(__bf16));
    __bf16* Wg3 = (__bf16*)carve((size_t)80  * 160 * 14 * sizeof(__bf16));
    __bf16* Wg4 = (__bf16*)carve((size_t)48  * 96  * 14 * sizeof(__bf16));

    // 1) FFT features
    {
        int n = B * 14;
        fft_feat_kernel<<<(n + 255) / 256, 256, 0, stream>>>(x, h0, B);
    }
    // 2) pack weights (bf16, padded)
    {
        int n1 = 80 * 512, n2 = 160 * 96, n3 = 80 * 160, n4 = 48 * 96;
        pack_w_kernel<<<(n1 + 255) / 256, 256, 0, stream>>>(k1_base, k1_spl, k1_scl, Wg1, 80, 80, 504, 512);
        pack_w_kernel<<<(n2 + 255) / 256, 256, 0, stream>>>(k2_base, k2_spl, k2_scl, Wg2, 160, 160, 80, 96);
        pack_w_kernel<<<(n3 + 255) / 256, 256, 0, stream>>>(k3_base, k3_spl, k3_scl, Wg3, 80, 80, 160, 160);
        pack_w_kernel<<<(n4 + 255) / 256, 256, 0, stream>>>(k4_base, k4_spl, k4_scl, Wg4, 40, 48, 80, 96);
    }
    // 3) fused KAN layers (WMMA), 32 rows per workgroup
    const int nrow = B / 32;
    kan_gemm_kernel<80> <<<nrow, 160, 0, stream>>>(h0, 504, 512, Wg1, hA, 80);
    kan_gemm_kernel<160><<<nrow, 320, 0, stream>>>(hA, 80,  96,  Wg2, hB, 160);
    kan_gemm_kernel<80> <<<nrow, 160, 0, stream>>>(hB, 160, 160, Wg3, hA, 80);
    kan_gemm_kernel<48> <<<nrow, 96,  0, stream>>>(hA, 80,  96,  Wg4, hB, 40);
    // 4) heads
    heads_kernel<<<(B + 127) / 128, 128, 0, stream>>>(hB, hW1, hb1, hW2, hb2, hW3, hb3, out, B);
}
